// EdgeDropViewLearner_86079734547103
// MI455X (gfx1250) — matrix operand order repaired and checked by reading, compiled, not run on hardware
//
#include <hip/hip_runtime.h>
#include <hip/hip_bf16.h>

typedef __attribute__((ext_vector_type(16))) __bf16 v16bf;
typedef __attribute__((ext_vector_type(8)))  __bf16 v8bf;
typedef __attribute__((ext_vector_type(4)))  __bf16 v4bf;
typedef __attribute__((ext_vector_type(8)))  float  v8f;
typedef __attribute__((ext_vector_type(4)))  float  f32x4;
typedef __attribute__((ext_vector_type(4)))  int    i32x4;

// Exact pointee type of the async-to-LDS builtin params (per hipcc diagnostic):
// '__attribute__((__vector_size__(4 * sizeof(int)))) int'
typedef int av4i __attribute__((vector_size(16)));
typedef __attribute__((address_space(1))) av4i* gv4i_ptr;   // global (AS1)
typedef __attribute__((address_space(3))) av4i* lv4i_ptr;   // LDS (AS3)

#define D_FEAT 128
#define HIDDEN 64
// K = 2*D_FEAT = 256 -> 8 chunks of 32; N = 64 -> 4 tiles of 16.
// W1 package: 1024 slots x 16 bf16 (hi) + same (lo) = 64 KB.
#define W1_SLOTS 1024
#define W1_ELEMS (W1_SLOTS * 16)   // 16384 per precision

#define HAS_ASYNC_LDS __has_builtin(__builtin_amdgcn_global_load_async_to_lds_b128)

// ---------------- Prepass A: node_emb fp32 -> bf16 hi/lo tables ------------
__global__ __launch_bounds__(256) void convert_emb_kernel(
    const float* __restrict__ emb, __bf16* __restrict__ hi,
    __bf16* __restrict__ lo, int n4)
{
    const int i = blockIdx.x * 256 + threadIdx.x;
    if (i >= n4) return;
    f32x4 v = ((const f32x4*)emb)[i];
    v4bf h, l;
    #pragma unroll
    for (int j = 0; j < 4; ++j) {
        __bf16 hj = (__bf16)v[j];
        h[j] = hj;
        l[j] = (__bf16)(v[j] - (float)hj);
    }
    ((v4bf*)hi)[i] = h;
    ((v4bf*)lo)[i] = l;
}

// ------- Prepass B: W1 fp32 -> bf16 hi/lo in WMMA B-matrix slot layout -----
// slot = ((t*8 + c)*32 + lane); element e of lane L maps to
//   K = 32*c + 8*(L>>4) + (e&7) + 16*(e>>3),  N = (L&15) + 16*t.
__global__ __launch_bounds__(256) void pack_w1_kernel(
    const float* __restrict__ W1, __bf16* __restrict__ pack)
{
    for (int s = threadIdx.x; s < W1_SLOTS; s += 256) {
        const int lane  = s & 31;
        const int tc    = s >> 5;
        const int t     = tc >> 3;
        const int c     = tc & 7;
        const int g     = lane >> 4;
        const int n     = (lane & 15) + 16 * t;
        const int kbase = 32 * c + 8 * g;
        #pragma unroll
        for (int e = 0; e < 16; ++e) {
            float  w  = W1[(kbase + (e & 7) + ((e >> 3) << 4)) * HIDDEN + n];
            __bf16 hi = (__bf16)w;
            pack[s * 16 + e]            = hi;
            pack[W1_ELEMS + s * 16 + e] = (__bf16)(w - (float)hi);
        }
    }
}

// ------------------------------ Main kernel --------------------------------
__global__ __launch_bounds__(256) void edge_mlp_wmma_kernel(
    const __bf16* __restrict__ emb_hi,
    const __bf16* __restrict__ emb_lo,
    const int*    __restrict__ src,
    const int*    __restrict__ dst,
    const __bf16* __restrict__ w1pack,
    const float*  __restrict__ b1,
    const float*  __restrict__ W2,
    const float*  __restrict__ b2,
    float*        __restrict__ out,
    int E)
{
    // hi at sB[0..16383], lo at sB[16384..32767]
    __shared__ __align__(16) __bf16 sB[2 * W1_ELEMS];

    const int tid = threadIdx.x;

    // Stage the prebuilt 64 KB W1 package into LDS via the async-to-LDS path.
    {
        char* gsrc = (char*)w1pack;
        char* ldst = (char*)&sB[0];
        #pragma unroll
        for (int it = 0; it < 16; ++it) {
            const int off = (it * 256 + tid) * 16;
#if HAS_ASYNC_LDS
            __builtin_amdgcn_global_load_async_to_lds_b128(
                (gv4i_ptr)(gsrc + off), (lv4i_ptr)(ldst + off), 0, 0);
#else
            i32x4 t = *(const i32x4*)(gsrc + off);
            *(i32x4*)(ldst + off) = t;
#endif
        }
#if HAS_ASYNC_LDS
  #if __has_builtin(__builtin_amdgcn_s_wait_asynccnt)
        __builtin_amdgcn_s_wait_asynccnt(0);
  #else
        asm volatile("s_wait_asynccnt 0" ::: "memory");
  #endif
#endif
    }
    __syncthreads();

    const int lane   = tid & 31;
    const int wave   = tid >> 5;
    const int g      = lane >> 4;   // lane-group: selects K sub-range / row half
    const int ln     = lane & 15;
    const int e_base = blockIdx.x * 128 + wave * 16;

    // Edge whose A-row this lane feeds (clamped gather; stores are predicated).
    int e_row = e_base + ln;
    if (e_row >= E) e_row = E - 1;
    const long src_off = (long)src[e_row] * D_FEAT;
    const long dst_off = (long)dst[e_row] * D_FEAT;

    // fp32 accumulators seeded with b1[n] (bias folded into the GEMM).
    v8f C[4];
    #pragma unroll
    for (int t = 0; t < 4; ++t) {
        const float bb = b1[ln + 16 * t];
        #pragma unroll
        for (int j = 0; j < 8; ++j) C[t][j] = bb;
    }

    #pragma unroll
    for (int c = 0; c < 8; ++c) {
        // A-matrix 16x32 bf16 per-lane layout: two contiguous 8-elem runs.
        // group 0 holds K = kb+{0..7,16..23}, group 1 holds K = kb+{8..15,24..31};
        // chunks 0-3 read the src embedding, 4-7 the dst embedding.
        const long row  = (c < 4) ? src_off : dst_off;
        const int  koff = (c & 3) * 32 + g * 8;
        const __bf16* ph = emb_hi + row + koff;
        const __bf16* pl = emb_lo + row + koff;
        v8bf h0 = *(const v8bf*)(ph);
        v8bf h1 = *(const v8bf*)(ph + 16);
        v8bf l0 = *(const v8bf*)(pl);
        v8bf l1 = *(const v8bf*)(pl + 16);
        v16bf ahi = __builtin_shufflevector(h0, h1, 0, 1, 2, 3, 4, 5, 6, 7,
                                            8, 9, 10, 11, 12, 13, 14, 15);
        v16bf alo = __builtin_shufflevector(l0, l1, 0, 1, 2, 3, 4, 5, 6, 7,
                                            8, 9, 10, 11, 12, 13, 14, 15);

        #pragma unroll
        for (int t = 0; t < 4; ++t) {
            const int slot = ((t * 8 + c) * 32 + lane) * 16;
            v16bf bhi = *(const v16bf*)&sB[slot];
            v16bf blo = *(const v16bf*)&sB[W1_ELEMS + slot];
            // split-bf16: hi*hi + hi*lo + lo*hi (fp32 accumulate, ~fp32 accuracy)
            C[t] = __builtin_amdgcn_wmma_f32_16x16x32_bf16(
                       false, ahi, false, bhi, (short)0, C[t], false, false);
            C[t] = __builtin_amdgcn_wmma_f32_16x16x32_bf16(
                       false, ahi, false, blo, (short)0, C[t], false, false);
            C[t] = __builtin_amdgcn_wmma_f32_16x16x32_bf16(
                       false, alo, false, bhi, (short)0, C[t], false, false);
        }
    }

    // Layer 2: logits[e] = sum_n relu(h[m][n]) * W2[n] + b2.
    // C/D layout: lane L, elem j -> M = j + 8*(L>>4), N = (L&15) + 16*t.
    float w2v[4];
    #pragma unroll
    for (int t = 0; t < 4; ++t) w2v[t] = W2[ln + 16 * t];
    const float b2s = b2[0];

    float p[8];
    #pragma unroll
    for (int j = 0; j < 8; ++j) {
        float acc = 0.0f;
        #pragma unroll
        for (int t = 0; t < 4; ++t) acc += fmaxf(C[t][j], 0.0f) * w2v[t];
        p[j] = acc;
    }
    // Reduce over N across the 16 lanes of each half-wave.
    #pragma unroll
    for (int j = 0; j < 8; ++j) {
        #pragma unroll
        for (int m = 1; m < 16; m <<= 1) p[j] += __shfl_xor(p[j], m, 32);
    }
    if (ln == 0) {
        const int e0 = e_base + g * 8;
        #pragma unroll
        for (int j = 0; j < 8; ++j)
            if (e0 + j < E) out[e0 + j] = p[j] + b2s;
    }
}

// ------------------------------- Launcher ----------------------------------
extern "C" void kernel_launch(void* const* d_in, const int* in_sizes, int n_in,
                              void* d_out, int out_size, void* d_ws, size_t ws_size,
                              hipStream_t stream) {
    const float* node_emb = (const float*)d_in[0];
    const int*   src      = (const int*)d_in[1];
    const int*   dst      = (const int*)d_in[2];
    const float* W1       = (const float*)d_in[3];
    const float* b1       = (const float*)d_in[4];
    const float* W2       = (const float*)d_in[5];
    const float* b2       = (const float*)d_in[6];
    float*       out      = (float*)d_out;

    const int nEmb = in_sizes[0];   // N_NODES * D_FEAT
    const int E    = in_sizes[1];   // number of edges

    // Workspace layout: emb_hi | emb_lo | w1pack(hi,lo)
    __bf16* emb_hi = (__bf16*)d_ws;
    __bf16* emb_lo = emb_hi + nEmb;
    __bf16* w1pack = emb_lo + nEmb;

    const int n4 = nEmb / 4;
    convert_emb_kernel<<<(n4 + 255) / 256, 256, 0, stream>>>(
        node_emb, emb_hi, emb_lo, n4);
    pack_w1_kernel<<<1, 256, 0, stream>>>(W1, w1pack);

    const int blocks = (E + 127) / 128;     // 128 edges per 256-thread block
    edge_mlp_wmma_kernel<<<blocks, 256, 0, stream>>>(
        emb_hi, emb_lo, src, dst, w1pack, b1, W2, b2, out, E);
}
